// GroupedQueryAttention_11656541241888
// MI455X (gfx1250) — matrix operand order repaired and checked by reading, not compiled, not run on hardware
//
#include <hip/hip_runtime.h>
#include <hip/hip_bf16.h>
#include <stdint.h>
#include <stddef.h>

typedef __attribute__((ext_vector_type(16))) __bf16 v16bf;
typedef __attribute__((ext_vector_type(8)))  float  v8f;

#define B_  2
#define S_  2048
#define D_  2048
#define H_  16
#define G_  4
#define HD_ 128

__device__ __forceinline__ unsigned short f2bf(float f) {
    union { float f; unsigned u; } a; a.f = f;
    unsigned u = a.u + 0x7FFFu + ((a.u >> 16) & 1u);
    return (unsigned short)(u >> 16);
}

union FragBf { uint4 q[2]; v16bf v; };

// 32-byte async copy global -> LDS (2x b128; IOFFSET applies to both addresses)
__device__ __forceinline__ void async_copy32B(const unsigned short* gsrc,
                                              unsigned short* ldst) {
    unsigned lds = (unsigned)(unsigned long long)(void*)ldst;   // low 32 bits = LDS offset
    unsigned long long ga = (unsigned long long)gsrc;
    asm volatile("global_load_async_to_lds_b128 %0, %1, off"
                 :: "v"(lds), "v"(ga) : "memory");
    asm volatile("global_load_async_to_lds_b128 %0, %1, off offset:16"
                 :: "v"(lds), "v"(ga) : "memory");
}

// ---------------- fp32 -> bf16 conversion ----------------
__global__ __launch_bounds__(256) void cvt_bf16(const float* __restrict__ in,
                                                unsigned short* __restrict__ out, int n) {
    int i = (blockIdx.x * 256 + threadIdx.x) * 4;
    if (i + 3 < n) {
        float4 f = *(const float4*)(in + i);
        ushort4 r;
        r.x = f2bf(f.x); r.y = f2bf(f.y); r.z = f2bf(f.z); r.w = f2bf(f.w);
        *(ushort4*)(out + i) = r;
    }
}

// ---------------- tiled bf16 WMMA GEMM: C = A @ W + bias ----------------
// MODE 0: bf16 row-major C.  MODE 1: KV split (K row-major, V transposed).
// MODE 2: fp32 row-major C (final output).
// A tile: double-buffered async global->LDS copies (ASYNCcnt path).
// W tile: register-transposed into LDS [n][k], pipelined one tile ahead.
template <int MODE>
__global__ __launch_bounds__(256) void gemm_bf16(
    const unsigned short* __restrict__ A,   // M x K
    const unsigned short* __restrict__ W,   // K x N
    const float* __restrict__ bias,         // N
    void* __restrict__ Cout,
    unsigned short* __restrict__ Kb,        // MODE 1: [B*G][S][HD]
    unsigned short* __restrict__ VTb,       // MODE 1: [B*G][HD][S]
    int M, int N, int K)
{
    constexpr int AS = 48;                        // padded LDS stride (bf16 units)
    __shared__ unsigned short As[2][128 * AS];    // A tile 128x32, double buffered
    __shared__ unsigned short Bs[2][128 * AS];    // W tile transposed [n][k], double buffered
    const int lane = threadIdx.x & 31;
    const int wave = threadIdx.x >> 5;
    const int wm = wave & 3, wn = wave >> 2;      // 4x2 wave grid
    const int m0 = blockIdx.y * 128, n0 = blockIdx.x * 128;
    const int kh = lane >> 4, ln = lane & 15;

    // cooperative-load thread mapping
    const int arow = threadIdx.x >> 1;            // 0..127
    const int acb  = (threadIdx.x & 1) * 16;      // 0 / 16
    const int kr   = threadIdx.x >> 3;            // 0..31
    const int nb   = (threadIdx.x & 7) * 16;      // 0..112
    const unsigned short* Arow = A + (size_t)(m0 + arow) * K + acb;
    const unsigned short* Wrow = W + (size_t)kr * N + n0 + nb;

    v8f acc[2][4];
    const v8f vzero = {0.f,0.f,0.f,0.f,0.f,0.f,0.f,0.f};
    #pragma unroll
    for (int mi = 0; mi < 2; ++mi)
        #pragma unroll
        for (int ni = 0; ni < 4; ++ni) acc[mi][ni] = vzero;

    // ---- prologue: fill buffer 0 with tile kk=0 ----
    async_copy32B(Arow, &As[0][arow * AS + acb]);
    {
        unsigned short tmp[16];
        *(uint4*)(tmp)     = ((const uint4*)Wrow)[0];
        *(uint4*)(tmp + 8) = ((const uint4*)Wrow)[1];
        #pragma unroll
        for (int i = 0; i < 16; ++i) Bs[0][(nb + i) * AS + kr] = tmp[i];
    }
    asm volatile("s_wait_asynccnt 0x0" ::: "memory");
    __syncthreads();

    for (int kk = 0; kk < K; kk += 32) {
        const int pb = (kk >> 5) & 1, qb = pb ^ 1;
        const bool more = (kk + 32) < K;

        // issue next A tile async copy into the other buffer
        if (more) async_copy32B(Arow + kk + 32, &As[qb][arow * AS + acb]);
        // fetch next W tile into registers
        unsigned short tmp[16];
        if (more) {
            const unsigned short* src = Wrow + (size_t)(kk + 32) * N;
            *(uint4*)(tmp)     = ((const uint4*)src)[0];
            *(uint4*)(tmp + 8) = ((const uint4*)src)[1];
        }

        // compute on buffer pb
        FragBf am[2], bn[4];
        #pragma unroll
        for (int mi = 0; mi < 2; ++mi) {          // A frag: lane row = ln, K split per half-wave
            int r = wm * 32 + mi * 16 + ln;
            am[mi].q[0] = *(const uint4*)(&As[pb][r * AS + kh * 8]);
            am[mi].q[1] = *(const uint4*)(&As[pb][r * AS + 16 + kh * 8]);
        }
        #pragma unroll
        for (int ni = 0; ni < 4; ++ni) {          // B frag: lane col = ln, K split per half-wave
            int c = wn * 64 + ni * 16 + ln;
            bn[ni].q[0] = *(const uint4*)(&Bs[pb][c * AS + kh * 8]);
            bn[ni].q[1] = *(const uint4*)(&Bs[pb][c * AS + 16 + kh * 8]);
        }
        #pragma unroll
        for (int mi = 0; mi < 2; ++mi)
            #pragma unroll
            for (int ni = 0; ni < 4; ++ni)
                acc[mi][ni] = __builtin_amdgcn_wmma_f32_16x16x32_bf16(
                    false, am[mi].v, false, bn[ni].v, (short)0, acc[mi][ni], false, false);

        // transpose-store next W tile into the other buffer
        if (more) {
            #pragma unroll
            for (int i = 0; i < 16; ++i) Bs[qb][(nb + i) * AS + kr] = tmp[i];
        }
        asm volatile("s_wait_asynccnt 0x0" ::: "memory");
        __syncthreads();
    }

    // epilogue: C layout = VGPR v -> row v (lanes<16) / v+8 (lanes>=16), col = ln
    #pragma unroll
    for (int mi = 0; mi < 2; ++mi)
        #pragma unroll
        for (int ni = 0; ni < 4; ++ni) {
            int n = n0 + wn * 64 + ni * 16 + ln;
            float bv = bias[n];
            #pragma unroll
            for (int v = 0; v < 8; ++v) {
                int m = m0 + wm * 32 + mi * 16 + v + (kh ? 8 : 0);
                float val = acc[mi][ni][v] + bv;
                if (MODE == 0) {
                    ((unsigned short*)Cout)[(size_t)m * N + n] = f2bf(val);
                } else if (MODE == 2) {
                    ((float*)Cout)[(size_t)m * N + n] = val;
                } else {
                    int b = m >> 11, s = m & (S_ - 1);   // m = b*S + s
                    int g = n >> 8,  r = n & 255;        // n = g*256 + r
                    if (r < HD_)
                        Kb[((size_t)(b * G_ + g) * S_ + s) * HD_ + r] = f2bf(val);
                    else
                        VTb[((size_t)(b * G_ + g) * HD_ + (r - HD_)) * S_ + s] = f2bf(val);
                }
            }
        }
}

// ---------------- flash attention (bf16 WMMA, fp32 online softmax) ----------------
__global__ __launch_bounds__(256) void attn_kernel(
    const unsigned short* __restrict__ Qb,   // [B*S][D]
    const unsigned short* __restrict__ Kb,   // [B*G][S][HD]
    const unsigned short* __restrict__ VTb,  // [B*G][HD][S]
    unsigned short* __restrict__ Ob)         // [B*S][D]
{
    __shared__ unsigned short P[8][16 * 32];     // per-wave P staging (C-layout -> A-layout)
    const int lane = threadIdx.x & 31;
    const int wave = threadIdx.x >> 5;
    const int kh = lane >> 4, ln = lane & 15;
    const int qt = blockIdx.x & 15;              // S/128 = 16 q-tiles
    const int h  = (blockIdx.x >> 4) & 15;
    const int b  = blockIdx.x >> 8;
    const int g  = h >> 2;                       // REP = 4
    const int q0 = qt * 128 + wave * 16;

    const unsigned short* Qbase = Qb  + (size_t)(b * S_ + q0) * D_ + h * HD_;
    const unsigned short* Kbase = Kb  + (size_t)(b * G_ + g) * S_ * HD_;
    const unsigned short* Vbase = VTb + (size_t)(b * G_ + g) * HD_ * S_;

    // Q fragments: 16 rows x 128 hd, 4 chunks of K=32 (held for whole kernel)
    FragBf qa[4];
    #pragma unroll
    for (int c = 0; c < 4; ++c) {
        const unsigned short* p = Qbase + (size_t)ln * D_ + c * 32 + kh * 8;
        qa[c].q[0] = *(const uint4*)(p);
        qa[c].q[1] = *(const uint4*)(p + 16);
    }

    const v8f vzero = {0.f,0.f,0.f,0.f,0.f,0.f,0.f,0.f};
    v8f o[8];
    float mrow[8], lrow[8];
    #pragma unroll
    for (int v = 0; v < 8; ++v) { mrow[v] = -1e30f; lrow[v] = 0.f; }
    #pragma unroll
    for (int cc = 0; cc < 8; ++cc) o[cc] = vzero;

    const float scale = 0.08838834764831845f;    // 1/sqrt(128)
    unsigned short* pw = &P[wave][0];

    for (int kt = 0; kt < S_ / 32; ++kt) {
        const int kbase = kt * 32;
        // scores: S[16q x 32k] = Q @ K^T, two 16-key B frags
        v8f sacc[2];
        #pragma unroll
        for (int j = 0; j < 2; ++j) {
            v8f a = vzero;
            const unsigned short* krow = Kbase + (size_t)(kbase + j * 16 + ln) * HD_;
            #pragma unroll
            for (int c = 0; c < 4; ++c) {
                FragBf kb;
                const unsigned short* p = krow + c * 32 + kh * 8;
                kb.q[0] = *(const uint4*)(p);
                kb.q[1] = *(const uint4*)(p + 16);
                a = __builtin_amdgcn_wmma_f32_16x16x32_bf16(
                        false, qa[c].v, false, kb.v, (short)0, a, false, false);
            }
            sacc[j] = a;
        }
        if (kt + 1 < S_ / 32)
            __builtin_prefetch(Kbase + (size_t)(kbase + 32 + ln) * HD_, 0, 1);

        // online softmax per row (rows live per-VGPR, cols across 16-lane halves)
        float fac[8];
        #pragma unroll
        for (int v = 0; v < 8; ++v) {
            float s0 = sacc[0][v] * scale, s1 = sacc[1][v] * scale;
            float mx = fmaxf(s0, s1);
            mx = fmaxf(mx, __shfl_xor(mx, 1));
            mx = fmaxf(mx, __shfl_xor(mx, 2));
            mx = fmaxf(mx, __shfl_xor(mx, 4));
            mx = fmaxf(mx, __shfl_xor(mx, 8));
            float mnew = fmaxf(mrow[v], mx);
            float p0 = __expf(s0 - mnew), p1 = __expf(s1 - mnew);
            float f  = __expf(mrow[v] - mnew);
            float rs = p0 + p1;
            rs += __shfl_xor(rs, 1); rs += __shfl_xor(rs, 2);
            rs += __shfl_xor(rs, 4); rs += __shfl_xor(rs, 8);
            lrow[v] = lrow[v] * f + rs;
            mrow[v] = mnew;
            fac[v]  = f;
            sacc[0][v] = p0; sacc[1][v] = p1;
        }
        #pragma unroll
        for (int cc = 0; cc < 8; ++cc)
            #pragma unroll
            for (int v = 0; v < 8; ++v) o[cc][v] *= fac[v];

        // stage P (C-layout) into LDS, reload as A-layout fragment
        #pragma unroll
        for (int j = 0; j < 2; ++j)
            #pragma unroll
            for (int v = 0; v < 8; ++v) {
                int row = v + (kh ? 8 : 0);
                pw[row * 32 + j * 16 + ln] = f2bf(sacc[j][v]);
            }
        asm volatile("s_wait_dscnt 0" ::: "memory");
        FragBf pa;
        {
            const unsigned short* pr = pw + ln * 32 + kh * 8;
            pa.q[0] = *(const uint4*)(pr);
            pa.q[1] = *(const uint4*)(pr + 16);
        }

        // O += P @ V, V read pre-transposed (contiguous along keys)
        #pragma unroll
        for (int cc = 0; cc < 8; ++cc) {
            FragBf vb;
            const unsigned short* p = Vbase + (size_t)(cc * 16 + ln) * S_ + kbase + kh * 8;
            vb.q[0] = *(const uint4*)(p);
            vb.q[1] = *(const uint4*)(p + 16);
            o[cc] = __builtin_amdgcn_wmma_f32_16x16x32_bf16(
                        false, pa.v, false, vb.v, (short)0, o[cc], false, false);
        }
    }

    // normalize and store O as bf16 rows of [B*S][D]
    #pragma unroll
    for (int v = 0; v < 8; ++v) {
        float inv = 1.0f / lrow[v];
        int qrow = q0 + v + (kh ? 8 : 0);
        unsigned short* orow = Ob + (size_t)(b * S_ + qrow) * D_ + h * HD_;
        #pragma unroll
        for (int cc = 0; cc < 8; ++cc)
            orow[cc * 16 + ln] = f2bf(o[cc][v] * inv);
    }
}

// ---------------- launcher ----------------
extern "C" void kernel_launch(void* const* d_in, const int* in_sizes, int n_in,
                              void* d_out, int out_size, void* d_ws, size_t ws_size,
                              hipStream_t stream) {
    const float* x   = (const float*)d_in[0];
    const float* Wq  = (const float*)d_in[1];
    const float* bq  = (const float*)d_in[2];
    const float* Wkv = (const float*)d_in[3];
    const float* bkv = (const float*)d_in[4];
    const float* Wo  = (const float*)d_in[5];
    const float* bo  = (const float*)d_in[6];
    float* out = (float*)d_out;

    const size_t M  = (size_t)B_ * S_;        // 4096
    char* p = (char*)d_ws;
    unsigned short* xb   = (unsigned short*)p; p += M * D_ * 2;
    unsigned short* Wqb  = (unsigned short*)p; p += (size_t)D_ * D_ * 2;
    unsigned short* Wkvb = (unsigned short*)p; p += (size_t)D_ * 1024 * 2;
    unsigned short* Wob  = (unsigned short*)p; p += (size_t)D_ * D_ * 2;
    unsigned short* Qb   = (unsigned short*)p; p += M * D_ * 2;
    unsigned short* Kb   = (unsigned short*)p; p += (size_t)B_ * G_ * S_ * HD_ * 2;
    unsigned short* VTb  = (unsigned short*)p; p += (size_t)B_ * G_ * S_ * HD_ * 2;
    unsigned short* Obuf = (unsigned short*)p; p += M * D_ * 2;

    auto cvt = [&](const float* src, unsigned short* dst, int n) {
        cvt_bf16<<<(n / 4 + 255) / 256, 256, 0, stream>>>(src, dst, n);
    };
    cvt(x,   xb,   (int)(M * D_));
    cvt(Wq,  Wqb,  D_ * D_);
    cvt(Wkv, Wkvb, D_ * 1024);
    cvt(Wo,  Wob,  D_ * D_);

    // Q = x @ Wq + bq                (bf16 row-major)
    gemm_bf16<0><<<dim3(D_ / 128, (int)(M / 128)), 256, 0, stream>>>(
        xb, Wqb, bq, (void*)Qb, nullptr, nullptr, (int)M, D_, D_);
    // KV = x @ Wkv + bkv             (K row-major, V transposed)
    gemm_bf16<1><<<dim3(1024 / 128, (int)(M / 128)), 256, 0, stream>>>(
        xb, Wkvb, bkv, nullptr, Kb, VTb, (int)M, 1024, D_);
    // flash attention
    attn_kernel<<<B_ * H_ * (S_ / 128), 256, 0, stream>>>(Qb, Kb, VTb, Obuf);
    // out = O @ Wo + bo              (fp32)
    gemm_bf16<2><<<dim3(D_ / 128, (int)(M / 128)), 256, 0, stream>>>(
        Obuf, Wob, bo, (void*)out, nullptr, nullptr, (int)M, D_, D_);
}
